// FGKAN_87789131531092
// MI455X (gfx1250) — matrix operand order repaired and checked by reading, not compiled, not run on hardware
//
#include <hip/hip_runtime.h>
#include <hip/hip_bf16.h>

// ---------------------------------------------------------------------------
// MI455X (gfx1250, wave32).  Heavy contractions (3x3 convs as implicit GEMM,
// token GEMMs, kNN Gram matrix) on v_wmma_f32_16x16x32_f16.
// Fragments are gathered directly from global memory with b128 vector loads
// (per-lane K-runs are contiguous in the wave32 WMMA layout) — no LDS staging.
// ---------------------------------------------------------------------------

typedef __attribute__((ext_vector_type(16))) _Float16 v16h;
typedef __attribute__((ext_vector_type(8)))  _Float16 h8v;
typedef __attribute__((ext_vector_type(8)))  float    v8f;
typedef __attribute__((ext_vector_type(4)))  float    f4v;

#define BATCH 64
#define HW    27
#define NPIX  729        // 27*27
#define DCH   64
#define NTOK  46656      // BATCH*NPIX

__device__ __forceinline__ v16h cat8(h8v lo, h8v hi) {
  return __builtin_shufflevector(lo, hi, 0, 1, 2, 3, 4, 5, 6, 7,
                                 8, 9, 10, 11, 12, 13, 14, 15);
}

__device__ __forceinline__ v8f wmma_f16(v16h a, v16h b, v8f c) {
  return __builtin_amdgcn_wmma_f32_16x16x32_f16(false, a, false, b, (short)0, c,
                                                false, false);
}

// ---------------------------------------------------------------------------
// Conv 3x3 SAME as implicit GEMM.  out[b, oc, p], p = y*27+x (raw, pre-BN).
// One wave computes a 16(oc) x 16(pixel) tile; K = Cin*9 in chunks of 32.
// A frag: per-lane contiguous weight runs (vector loads on full chunks).
// B frag: per-lane im2col gather straight into the fragment (no LDS).
// grid = (46 pixel tiles, Cout/16, B), block = 32.
// ---------------------------------------------------------------------------
__global__ void conv3x3_wmma(const float* __restrict__ in,
                             const float* __restrict__ wgt,
                             const float* __restrict__ bias,
                             float* __restrict__ out,
                             int Cin, int Cout) {
  const int pt   = blockIdx.x;
  const int ot   = blockIdx.y;
  const int b    = blockIdx.z;
  const int lane = threadIdx.x;
  const int K    = Cin * 9;
  const int p0   = pt * 16;
  const int oc0  = ot * 16;
  const int m    = lane & 15;
  const int hi   = lane >> 4;

  const int  p  = p0 + m;               // this lane's B column (pixel)
  const bool pv = (p < NPIX);
  const int  y  = p / HW, x = p - y * HW;
  const float* inb  = in + (size_t)b * Cin * NPIX;
  const float* wrow = wgt + (size_t)(oc0 + m) * K;   // this lane's A row

  v8f acc = {};
  for (int k0 = 0; k0 < K; k0 += 32) {
    // ---- A fragment: row m, K run [k0+hi*16, +16) ----
    v16h a;
    if (k0 + 32 <= K) {
      const f4v* wp = (const f4v*)(wrow + k0);
      f4v w0 = wp[hi * 4];
      f4v w1 = wp[hi * 4 + 1];
      f4v w2 = wp[hi * 4 + 2];
      f4v w3 = wp[hi * 4 + 3];
#pragma unroll
      for (int h = 0; h < 4; ++h) {
        a[h]      = (_Float16)w0[h];
        a[h + 4]  = (_Float16)w1[h];
        a[h + 8]  = (_Float16)w2[h];
        a[h + 12] = (_Float16)w3[h];
      }
    } else {  // guarded scalar tail (only conv1: K=1800)
#pragma unroll
      for (int h = 0; h < 16; ++h) {
        int k = k0 + hi * 16 + h;
        a[h] = (_Float16)((k < K) ? wrow[k] : 0.f);
      }
    }
    // ---- B fragment: col = p, K = k0 + hi*16 + h (im2col gather) ----
    v16h bf;
    {
      int kbase = k0 + hi * 16;
      int ic = kbase / 9;
      int r9 = kbase - ic * 9;
#pragma unroll
      for (int h = 0; h < 16; ++h) {
        float v = 0.f;
        int k = kbase + h;
        if (pv && k < K) {
          int ky = r9 / 3, kx = r9 - ky * 3;
          int iy = y + ky - 1, ix = x + kx - 1;
          if (iy >= 0 && iy < HW && ix >= 0 && ix < HW)
            v = inb[(size_t)ic * NPIX + iy * HW + ix];
        }
        bf[h] = (_Float16)v;
        if (++r9 == 9) { r9 = 0; ++ic; }
      }
    }
    if (k0 + 32 < K) __builtin_prefetch(wrow + k0 + 32, 0, 1);
    acc = wmma_f16(a, bf, acc);
  }
  // store D: M = r + hi*8, N = m
  if (pv) {
#pragma unroll
    for (int r = 0; r < 8; ++r) {
      int oc = oc0 + r + hi * 8;
      out[((size_t)(b * Cout + oc)) * NPIX + p] = acc[r] + bias[oc];
    }
  }
}

// ---------------------------------------------------------------------------
// BatchNorm (training mode) over (B, spatial) for channel-major [B, C, NPIX].
// ---------------------------------------------------------------------------
__global__ void bn_stats_cmajor(const float* __restrict__ x, float* mean,
                                float* var, int C) {
  const int c = blockIdx.x, tid = threadIdx.x;
  const int total = BATCH * NPIX;
  float s = 0.f, ss = 0.f;
  for (int i = tid; i < total; i += blockDim.x) {
    int b = i / NPIX, n = i - b * NPIX;
    float v = x[((size_t)(b * C + c)) * NPIX + n];
    s += v; ss += v * v;
  }
  __shared__ float rs[256], rss[256];
  rs[tid] = s; rss[tid] = ss;
  __syncthreads();
  for (int w = 128; w > 0; w >>= 1) {
    if (tid < w) { rs[tid] += rs[tid + w]; rss[tid] += rss[tid + w]; }
    __syncthreads();
  }
  if (tid == 0) {
    float mm = rs[0] / total;
    mean[c] = mm;
    var[c] = rss[0] / total - mm * mm;
  }
}

__global__ void bn_apply_cmajor(float* __restrict__ x, const float* mean,
                                const float* var, const float* gg,
                                const float* bb, int C, long total, int relu) {
  long i = (long)blockIdx.x * blockDim.x + threadIdx.x;
  if (i >= total) return;
  int c = (int)((i / NPIX) % C);
  float v = (x[i] - mean[c]) * rsqrtf(var[c] + 1e-5f) * gg[c] + bb[c];
  x[i] = relu ? fmaxf(v, 0.f) : v;
}

// ---------------------------------------------------------------------------
// Tokenize: [B, 64, NPIX] -> th f16 [B*NPIX, 64], xx = ||t||^2 per token.
// ---------------------------------------------------------------------------
__global__ void tokenize(const float* __restrict__ h3, _Float16* __restrict__ th,
                         float* __restrict__ xx) {
  int i = blockIdx.x * blockDim.x + threadIdx.x;
  if (i >= NTOK) return;
  int b = i / NPIX, n = i - b * NPIX;
  float s = 0.f;
#pragma unroll 8
  for (int d = 0; d < DCH; ++d) {
    float v = h3[((size_t)(b * DCH + d)) * NPIX + n];
    th[(size_t)i * DCH + d] = (_Float16)v;
    s += v * v;
  }
  xx[i] = s;
}

// ---- weight transpose + f16 convert: Wt[n][k] = W[k][n], 64x64 -------------
__global__ void transposeW_f16(const float* __restrict__ W,
                               _Float16* __restrict__ Wt) {
  int i = blockIdx.x * blockDim.x + threadIdx.x;
  if (i >= DCH * DCH) return;
  int k = i >> 6, n = i & 63;
  Wt[n * DCH + k] = (_Float16)W[k * DCH + n];
}

// ---------------------------------------------------------------------------
// Token GEMM: out[NTOK,64] = th[NTOK,64] @ W[64,64].
// B source is pre-transposed f16 (n-major) -> both frags are pure b128 loads.
// grid = (NTOK/16, 4), block = 32 (one wave per 16x16 tile).
// ---------------------------------------------------------------------------
__global__ void gemm_tok64(const _Float16* __restrict__ th,
                           const _Float16* __restrict__ Wt,
                           float* __restrict__ out) {
  const int mt = blockIdx.x, nt = blockIdx.y, lane = threadIdx.x;
  const int l = lane & 15, hi = lane >> 4;
  const h8v* ar = (const h8v*)(th + (size_t)(mt * 16 + l) * DCH);
  const h8v* br = (const h8v*)(Wt + (size_t)(nt * 16 + l) * DCH);
  v16h a0 = cat8(ar[hi * 2],     ar[hi * 2 + 1]);      // K chunk 0
  v16h a1 = cat8(ar[4 + hi * 2], ar[5 + hi * 2]);      // K chunk 1
  v16h b0 = cat8(br[hi * 2],     br[hi * 2 + 1]);
  v16h b1 = cat8(br[4 + hi * 2], br[5 + hi * 2]);
  v8f acc = {};
  acc = wmma_f16(a0, b0, acc);
  acc = wmma_f16(a1, b1, acc);
#pragma unroll
  for (int r = 0; r < 8; ++r)
    out[(size_t)(mt * 16 + r + hi * 8) * DCH + nt * 16 + l] = acc[r];
}

// ---------------------------------------------------------------------------
// kNN: Gram tiles via WMMA (A rows and B cols both direct b128 loads from th),
// distance tile through LDS for the C-layout -> per-row handoff, register
// insertion-sorted top-9 per row.  idx[b,n,0..7] = neighbors (skip self).
// grid = (46 row tiles, B), block = 32.
// ---------------------------------------------------------------------------
__global__ void knn_topk(const _Float16* __restrict__ th,
                         const float* __restrict__ xx, int* __restrict__ idx) {
  const int rt = blockIdx.x, b = blockIdx.y, lane = threadIdx.x;
  const int r0 = rt * 16;
  const int m = lane & 15, hi = lane >> 4;
  __shared__ float dts[16][17];

  int rowa = r0 + m; if (rowa > NPIX - 1) rowa = NPIX - 1;   // clamp
  const h8v* ar = (const h8v*)(th + (size_t)(b * NPIX + rowa) * DCH);
  v16h a0 = cat8(ar[hi * 2],     ar[hi * 2 + 1]);
  v16h a1 = cat8(ar[4 + hi * 2], ar[5 + hi * 2]);

  float bd[9]; int bi[9];
#pragma unroll
  for (int j = 0; j < 9; ++j) { bd[j] = 3.4e38f; bi[j] = 0; }
  const int myrow = r0 + lane;
  const bool rowv = (lane < 16) && (myrow < NPIX);
  const float xr = rowv ? xx[b * NPIX + myrow] : 0.f;

  for (int ct = 0; ct < 46; ++ct) {
    const int c0 = ct * 16;
    int col = c0 + m; if (col > NPIX - 1) col = NPIX - 1;    // clamp
    const h8v* br = (const h8v*)(th + (size_t)(b * NPIX + col) * DCH);
    v16h b0 = cat8(br[hi * 2],     br[hi * 2 + 1]);
    v16h b1 = cat8(br[4 + hi * 2], br[5 + hi * 2]);
    v8f acc = {};
    acc = wmma_f16(a0, b0, acc);
    acc = wmma_f16(a1, b1, acc);
#pragma unroll
    for (int r = 0; r < 8; ++r) dts[r + hi * 8][m] = acc[r];
    __syncthreads();
    if (rowv) {
      for (int c = 0; c < 16; ++c) {
        int cc = c0 + c;
        if (cc >= NPIX) break;
        float d = fmaxf(xr - 2.f * dts[lane][c] + xx[b * NPIX + cc], 0.f);
        if (d < bd[8]) {
          int j = 8;
          while (j > 0 && bd[j - 1] > d) {
            bd[j] = bd[j - 1]; bi[j] = bi[j - 1]; --j;
          }
          bd[j] = d; bi[j] = cc;
        }
      }
    }
    __syncthreads();
  }
  if (rowv) {
#pragma unroll
    for (int j = 0; j < 8; ++j)
      idx[(size_t)(b * NPIX + myrow) * 8 + j] = bi[j + 1];
  }
}

// ---------------------------------------------------------------------------
// GKAN: g = hs + (u[n] + c2 * sum_nb u[m]) + bias, then rational activation.
// top-k gives 8 distinct neighbors -> deg = 8+1e-6, c2 = dinv[n]*dinv[m].
// grid = NTOK, block = 64.
// ---------------------------------------------------------------------------
__global__ void gkan_agg(const float* __restrict__ hs, const float* __restrict__ u,
                         const int* __restrict__ idx, const float* __restrict__ gbias,
                         const float* __restrict__ ka, const float* __restrict__ kb,
                         float* __restrict__ g) {
  const int t = blockIdx.x, d = threadIdx.x;
  const int b = t / NPIX;
  const float c2 = 1.0f / (8.0f + 1e-6f);
  float acc = u[(size_t)t * DCH + d];
#pragma unroll
  for (int j = 0; j < 8; ++j) {
    int mm = idx[(size_t)t * 8 + j];
    acc += c2 * u[(size_t)(b * NPIX + mm) * DCH + d];
  }
  float gv = hs[(size_t)t * DCH + d] + acc + gbias[d];
  float g2 = gv * gv;
  float num = ka[d * 3 + 0] + ka[d * 3 + 1] * gv + ka[d * 3 + 2] * g2;
  float den = 1.0f + fabsf(kb[d * 2 + 0] * gv + kb[d * 2 + 1] * g2);
  g[(size_t)t * DCH + d] = num / (den + 1e-8f);
}

// ---- BatchNorm over tokens for token-major [NTOK, 64] ----------------------
__global__ void bn_stats_tmajor(const float* __restrict__ x, float* mean,
                                float* var) {
  const int d = blockIdx.x, tid = threadIdx.x;
  float s = 0.f, ss = 0.f;
  for (int i = tid; i < NTOK; i += blockDim.x) {
    float v = x[(size_t)i * DCH + d];
    s += v; ss += v * v;
  }
  __shared__ float rs[256], rss[256];
  rs[tid] = s; rss[tid] = ss;
  __syncthreads();
  for (int w = 128; w > 0; w >>= 1) {
    if (tid < w) { rs[tid] += rs[tid + w]; rss[tid] += rss[tid + w]; }
    __syncthreads();
  }
  if (tid == 0) {
    float mm = rs[0] / NTOK;
    mean[d] = mm;
    var[d] = rss[0] / NTOK - mm * mm;
  }
}

__global__ void bn_apply_tmajor(float* __restrict__ x, const float* mean,
                                const float* var, const float* gg,
                                const float* bb) {
  long i = (long)blockIdx.x * blockDim.x + threadIdx.x;
  if (i >= (long)NTOK * DCH) return;
  int d = (int)(i & (DCH - 1));
  x[i] = (x[i] - mean[d]) * rsqrtf(var[d] + 1e-5f) * gg[d] + bb[d];
}

// ---------------------------------------------------------------------------
// Fuzzy attention: RBF memberships -> 64->16->64 MLP -> residual gate.
// grid = NTOK, block = 64.
// ---------------------------------------------------------------------------
__global__ void fuzzy_attn(const float* __restrict__ gin,
                           const float* __restrict__ centers,
                           const float* __restrict__ gamma,
                           const float* __restrict__ w1, const float* __restrict__ b1,
                           const float* __restrict__ w2, const float* __restrict__ b2,
                           float* __restrict__ gout) {
  const int t = blockIdx.x, d = threadIdx.x;
  __shared__ float rbf[64];
  __shared__ float a1[16];
  float gv = gin[(size_t)t * DCH + d];
  float s = 0.f;
#pragma unroll
  for (int f = 0; f < 9; ++f) {
    float diff = gv - centers[d * 9 + f];
    s += expf(-fabsf(gamma[d * 9 + f]) * diff * diff) + 1e-10f;
  }
  rbf[d] = s;
  __syncthreads();
  if (d < 16) {
    float a = b1[d];
#pragma unroll 8
    for (int k = 0; k < 64; ++k) a += rbf[k] * w1[d * 64 + k];
    a1[d] = fmaxf(a, 0.f);
  }
  __syncthreads();
  float a = b2[d];
#pragma unroll
  for (int k = 0; k < 16; ++k) a += a1[k] * w2[d * 16 + k];
  float attn = 1.f / (1.f + expf(-a));
  gout[(size_t)t * DCH + d] = gv * attn + gv;
}

// ---- global pool + classifier ---------------------------------------------
__global__ void pool_kernel(const float* __restrict__ g, float* __restrict__ p) {
  const int b = blockIdx.x, d = threadIdx.x;
  float s = 0.f, mx = -3.4e38f;
  for (int n = 0; n < NPIX; ++n) {
    float v = g[(size_t)(b * NPIX + n) * DCH + d];
    s += v; mx = fmaxf(mx, v);
  }
  p[b * DCH + d] = 0.5f * (s / (float)NPIX) + 0.5f * mx;
}

__global__ void fc1_kernel(const float* __restrict__ p, const float* __restrict__ w,
                           const float* __restrict__ bias, float* __restrict__ c) {
  const int b = blockIdx.x, j = threadIdx.x;  // block=128
  float s = bias[j];
#pragma unroll 8
  for (int k = 0; k < 64; ++k) s += p[b * 64 + k] * w[j * 64 + k];
  c[b * 128 + j] = s;
}

__global__ void cls_bn_relu(float* __restrict__ c, const float* __restrict__ g,
                            const float* __restrict__ bta) {
  const int j = threadIdx.x;  // single block of 128
  float s = 0.f, ss = 0.f;
  for (int b = 0; b < BATCH; ++b) {
    float v = c[b * 128 + j];
    s += v; ss += v * v;
  }
  float m = s / (float)BATCH;
  float var = ss / (float)BATCH - m * m;
  float inv = rsqrtf(var + 1e-5f);
  for (int b = 0; b < BATCH; ++b) {
    float v = (c[b * 128 + j] - m) * inv * g[j] + bta[j];
    c[b * 128 + j] = fmaxf(v, 0.f);
  }
}

__global__ void fc2_kernel(const float* __restrict__ c, const float* __restrict__ w,
                           const float* __restrict__ bias, float* __restrict__ out) {
  const int b = blockIdx.x, j = threadIdx.x;  // block=16
  float s = bias[j];
#pragma unroll 8
  for (int k = 0; k < 128; ++k) s += c[b * 128 + k] * w[j * 128 + k];
  out[b * 16 + j] = s;
}

// ---------------------------------------------------------------------------
extern "C" void kernel_launch(void* const* d_in, const int* in_sizes, int n_in,
                              void* d_out, int out_size, void* d_ws, size_t ws_size,
                              hipStream_t stream) {
  const float* x        = (const float*)d_in[0];
  const float* conv1_w  = (const float*)d_in[1];
  const float* conv1_b  = (const float*)d_in[2];
  const float* bn1_g    = (const float*)d_in[3];
  const float* bn1_b    = (const float*)d_in[4];
  const float* conv2_w  = (const float*)d_in[5];
  const float* conv2_b  = (const float*)d_in[6];
  const float* bn2_g    = (const float*)d_in[7];
  const float* bn2_b    = (const float*)d_in[8];
  const float* conv3_w  = (const float*)d_in[9];
  const float* conv3_b  = (const float*)d_in[10];
  const float* bn3_g    = (const float*)d_in[11];
  const float* bn3_b    = (const float*)d_in[12];
  const float* w_self   = (const float*)d_in[13];
  const float* w_nb     = (const float*)d_in[14];
  const float* gkan_bias= (const float*)d_in[15];
  const float* kan_a    = (const float*)d_in[16];
  const float* kan_b    = (const float*)d_in[17];
  const float* gkan_bn_g= (const float*)d_in[18];
  const float* gkan_bn_b= (const float*)d_in[19];
  const float* centers  = (const float*)d_in[20];
  const float* gamma    = (const float*)d_in[21];
  const float* attn_w1  = (const float*)d_in[22];
  const float* attn_b1  = (const float*)d_in[23];
  const float* attn_w2  = (const float*)d_in[24];
  const float* attn_b2  = (const float*)d_in[25];
  const float* fa_bn_g  = (const float*)d_in[26];
  const float* fa_bn_b  = (const float*)d_in[27];
  const float* fc1_w    = (const float*)d_in[28];
  const float* fc1_b    = (const float*)d_in[29];
  const float* cls_bn_g = (const float*)d_in[30];
  const float* cls_bn_b = (const float*)d_in[31];
  const float* fc2_w    = (const float*)d_in[32];
  const float* fc2_b    = (const float*)d_in[33];

  // ---- workspace carve-out -------------------------------------------------
  char* base = (char*)d_ws;
  size_t cur = 0;
  auto carve = [&](size_t bytes) -> char* {
    char* pp = base + cur;
    cur += (bytes + 255) & ~(size_t)255;
    return pp;
  };
  float*     h1   = (float*)carve((size_t)BATCH * 32 * NPIX * 4);   // 6 MB
  float*     h2   = (float*)carve((size_t)BATCH * 64 * NPIX * 4);   // 12 MB
  float*     h3   = (float*)carve((size_t)BATCH * 64 * NPIX * 4);   // 12 MB
  _Float16*  th   = (_Float16*)carve((size_t)NTOK * DCH * 2);       // 6 MB
  float*     xx   = (float*)carve((size_t)NTOK * 4);
  float*     hs   = (float*)carve((size_t)NTOK * DCH * 4);          // 12 MB
  float*     u    = (float*)carve((size_t)NTOK * DCH * 4);          // 12 MB
  int*       nidx = (int*)carve((size_t)NTOK * 8 * 4);              // 1.5 MB
  float*     mean = (float*)carve(128 * 4);
  float*     var  = (float*)carve(128 * 4);
  _Float16*  wtA  = (_Float16*)carve((size_t)DCH * DCH * 2);        // w_self^T f16
  _Float16*  wtB  = (_Float16*)carve((size_t)DCH * DCH * 2);        // w_nb^T f16
  float*     p    = (float*)carve((size_t)BATCH * DCH * 4);
  float*     c1   = (float*)carve((size_t)BATCH * 128 * 4);
  float*     g    = h2;   // reuse: h2 dead after conv3
  float*     gout = hs;   // reuse: hs dead after gkan_agg

  // ---- CNN encoder ---------------------------------------------------------
  conv3x3_wmma<<<dim3(46, 2, BATCH), 32, 0, stream>>>(x, conv1_w, conv1_b, h1, 200, 32);
  bn_stats_cmajor<<<32, 256, 0, stream>>>(h1, mean, var, 32);
  {
    long tot = (long)BATCH * 32 * NPIX;
    bn_apply_cmajor<<<(int)((tot + 255) / 256), 256, 0, stream>>>(h1, mean, var, bn1_g, bn1_b, 32, tot, 1);
  }
  conv3x3_wmma<<<dim3(46, 4, BATCH), 32, 0, stream>>>(h1, conv2_w, conv2_b, h2, 32, 64);
  bn_stats_cmajor<<<64, 256, 0, stream>>>(h2, mean, var, 64);
  {
    long tot = (long)BATCH * 64 * NPIX;
    bn_apply_cmajor<<<(int)((tot + 255) / 256), 256, 0, stream>>>(h2, mean, var, bn2_g, bn2_b, 64, tot, 1);
  }
  conv3x3_wmma<<<dim3(46, 4, BATCH), 32, 0, stream>>>(h2, conv3_w, conv3_b, h3, 64, 64);
  bn_stats_cmajor<<<64, 256, 0, stream>>>(h3, mean, var, 64);
  {
    long tot = (long)BATCH * 64 * NPIX;
    bn_apply_cmajor<<<(int)((tot + 255) / 256), 256, 0, stream>>>(h3, mean, var, bn3_g, bn3_b, 64, tot, 1);
  }

  // ---- tokenize + token GEMMs + kNN ---------------------------------------
  tokenize<<<(NTOK + 255) / 256, 256, 0, stream>>>(h3, th, xx);
  transposeW_f16<<<(DCH * DCH + 255) / 256, 256, 0, stream>>>(w_self, wtA);
  transposeW_f16<<<(DCH * DCH + 255) / 256, 256, 0, stream>>>(w_nb, wtB);
  gemm_tok64<<<dim3(NTOK / 16, 4), 32, 0, stream>>>(th, wtA, hs);
  gemm_tok64<<<dim3(NTOK / 16, 4), 32, 0, stream>>>(th, wtB, u);
  knn_topk<<<dim3(46, BATCH), 32, 0, stream>>>(th, xx, nidx);

  // ---- GKAN + BN -----------------------------------------------------------
  gkan_agg<<<NTOK, 64, 0, stream>>>(hs, u, nidx, gkan_bias, kan_a, kan_b, g);
  bn_stats_tmajor<<<64, 256, 0, stream>>>(g, mean, var);
  bn_apply_tmajor<<<(int)(((long)NTOK * DCH + 255) / 256), 256, 0, stream>>>(g, mean, var, gkan_bn_g, gkan_bn_b);

  // ---- fuzzy attention + BN ------------------------------------------------
  fuzzy_attn<<<NTOK, 64, 0, stream>>>(g, centers, gamma, attn_w1, attn_b1, attn_w2, attn_b2, gout);
  bn_stats_tmajor<<<64, 256, 0, stream>>>(gout, mean, var);
  bn_apply_tmajor<<<(int)(((long)NTOK * DCH + 255) / 256), 256, 0, stream>>>(gout, mean, var, fa_bn_g, fa_bn_b);

  // ---- pool + classifier ---------------------------------------------------
  pool_kernel<<<BATCH, 64, 0, stream>>>(gout, p);
  fc1_kernel<<<BATCH, 128, 0, stream>>>(p, fc1_w, fc1_b, c1);
  cls_bn_relu<<<1, 128, 0, stream>>>(c1, cls_bn_g, cls_bn_b);
  fc2_kernel<<<BATCH, 16, 0, stream>>>(c1, fc2_w, fc2_b, (float*)d_out);
}
